// KAGNMoE_72550587564099
// MI455X (gfx1250) — matrix operand order repaired and checked by reading, not compiled, not run on hardware
//
#include <hip/hip_runtime.h>
#include <hip/hip_bf16.h>

typedef __attribute__((ext_vector_type(16))) __bf16 v16bf;
typedef __attribute__((ext_vector_type(8)))  __bf16 v8bf;
typedef __attribute__((ext_vector_type(8)))  float  v8f;
typedef __attribute__((ext_vector_type(4)))  int    v4i;

#define NB   8     // batch
#define NE   8     // experts
#define CIN  64
#define COUT 128
#define HH   64
#define WW   64
#define HW   4096
#define KB   256   // basis channels = CIN*(DEGREE+1)
#define NPAIR 16   // NB * TOPK
#define PIXSTR 264 // padded bf16 per pixel in LDS window (bank-conflict-free)

#if __has_builtin(__builtin_amdgcn_global_load_async_to_lds_b128)
#define HAS_ASYNC_LDS 1
typedef __attribute__((address_space(1))) v4i gas_v4i;
typedef __attribute__((address_space(3))) v4i las_v4i;
#else
#define HAS_ASYNC_LDS 0
#endif

// ---------------- workspace layout (bytes) ----------------
#define OFF_MEANS 0u                      // 512 f32
#define OFF_GATES 2048u                   // 64 f32
#define OFF_TKI   2304u                   // 16 int
#define OFF_TKG   2368u                   // 16 f32
#define OFF_WPACK 4096u                   // 8*8*9*8*32*16 bf16 = 4,718,592 B
#define OFF_BASIS (4096u + 4718592u)      // 16*4096*256 bf16   = 33,554,432 B

// ---------------------------------------------------------------------------
// 1) per-(b,c) spatial mean
// ---------------------------------------------------------------------------
__global__ void k_mean(const float* __restrict__ x, float* __restrict__ means) {
    __shared__ float red[256];
    int bc = blockIdx.x;                       // b*CIN + c
    const float* p = x + (size_t)bc * HW;
    float s = 0.f;
    for (int i = threadIdx.x; i < HW; i += 256) s += p[i];
    red[threadIdx.x] = s;
    __syncthreads();
    for (int off = 128; off > 0; off >>= 1) {
        if ((int)threadIdx.x < off) red[threadIdx.x] += red[threadIdx.x + off];
        __syncthreads();
    }
    if (threadIdx.x == 0) means[bc] = red[0] * (1.0f / HW);
}

// ---------------------------------------------------------------------------
// 2) gating: softmax, top-2, gates, cv_squared loss
// ---------------------------------------------------------------------------
__global__ void k_gate(const float* __restrict__ means, const float* __restrict__ wg,
                       float* __restrict__ gates, int* __restrict__ tki,
                       float* __restrict__ tkg, float* __restrict__ loss_out) {
    __shared__ float logits[NB][NE];
    __shared__ float sg[NB][NE];
    int t = threadIdx.x;
    if (t < NB * NE) {
        int b = t / NE, e = t % NE;
        float s = 0.f;
        for (int c = 0; c < CIN; ++c) s += means[b * CIN + c] * wg[c * NE + e];
        logits[b][e] = s;
    }
    __syncthreads();
    if (t < NB) {
        int b = t;
        float mx = logits[b][0];
        for (int e = 1; e < NE; ++e) mx = fmaxf(mx, logits[b][e]);
        float pr[NE]; float den = 0.f;
        for (int e = 0; e < NE; ++e) { pr[e] = __expf(logits[b][e] - mx); den += pr[e]; }
        float inv = 1.0f / den;
        for (int e = 0; e < NE; ++e) pr[e] *= inv;
        int i0 = 0;
        for (int e = 1; e < NE; ++e) if (pr[e] > pr[i0]) i0 = e;
        int i1 = (i0 == 0) ? 1 : 0;
        for (int e = 0; e < NE; ++e) if (e != i0 && pr[e] > pr[i1]) i1 = e;
        float v0 = pr[i0], v1 = pr[i1];
        float s2 = v0 + v1 + 1e-6f;
        float g0 = v0 / s2, g1 = v1 / s2;
        for (int e = 0; e < NE; ++e) sg[b][e] = 0.f;
        sg[b][i0] = g0; sg[b][i1] = g1;
        tki[b * 2 + 0] = i0; tki[b * 2 + 1] = i1;
        tkg[b * 2 + 0] = g0; tkg[b * 2 + 1] = g1;
        for (int e = 0; e < NE; ++e) gates[b * NE + e] = sg[b][e];
    }
    __syncthreads();
    if (t == 0) {
        float imp[NE], ld[NE];
        for (int e = 0; e < NE; ++e) {
            float si = 0.f, sl = 0.f;
            for (int b = 0; b < NB; ++b) {
                si += sg[b][e];
                sl += (sg[b][e] > 0.f) ? 1.f : 0.f;
            }
            imp[e] = si; ld[e] = sl;
        }
        float loss = 0.f;
        for (int k = 0; k < 2; ++k) {
            const float* v = k ? ld : imp;
            float m = 0.f;
            for (int e = 0; e < NE; ++e) m += v[e];
            m *= (1.0f / NE);
            float var = 0.f;
            for (int e = 0; e < NE; ++e) { float d = v[e] - m; var += d * d; }
            var *= (1.0f / (NE - 1));                 // ddof=1
            loss += var / (m * m + 1e-10f);
        }
        *loss_out = loss * 0.01f;
    }
}

// ---------------------------------------------------------------------------
// 3) repack poly_weights (E,128,256,3,3) f32 -> bf16 in WMMA A-matrix order
//    lane<16 holds K {0..7,16..23}, lane>=16 holds K {8..15,24..31}
// ---------------------------------------------------------------------------
__global__ void k_pack(const float* __restrict__ pw, __hip_bfloat16* __restrict__ wpack) {
    int id = blockIdx.x * 256 + threadIdx.x;   // 0 .. 8*8*9*8*32-1 = 18431
    int lane = id & 31; int t = id >> 5;
    int kc  = t & 7;  t >>= 3;
    int tap = t % 9;  t /= 9;
    int mt  = t & 7;  int e = t >> 3;
    int m = mt * 16 + (lane & 15);
    int khalf = (lane >= 16) ? 8 : 0;
    __hip_bfloat16* dst = wpack + (size_t)id * 16;
#pragma unroll
    for (int i = 0; i < 16; ++i) {
        int klocal = i + ((i >= 8) ? 8 : 0) + khalf;
        int ch = kc * 32 + klocal;
        float v = pw[(((size_t)(e * COUT + m)) * KB + ch) * 9 + tap];
        dst[i] = __float2bfloat16(v);
    }
}

// ---------------------------------------------------------------------------
// 4) basis: tanh -> Gram recurrence -> SiLU -> bf16, layout [pair][pixel][ch]
// ---------------------------------------------------------------------------
__global__ void k_basis(const float* __restrict__ x, const float* __restrict__ beta,
                        const int* __restrict__ tki, __hip_bfloat16* __restrict__ basis) {
    __shared__ float tsh[64][65];              // padded: no bank conflicts
    int pair = blockIdx.x;                     // b*2 + kk
    int b = pair >> 1;
    int e = tki[pair];
    int pix0 = blockIdx.y * 64;
    float bw1 = beta[e * 4 + 1];
    float bw2 = beta[e * 4 + 2];
    const float* xb = x + (size_t)b * CIN * HW;
#pragma unroll
    for (int i = 0; i < 16; ++i) {
        int idx = threadIdx.x + i * 256;       // 0..4095
        int c = idx >> 6, pix = idx & 63;
        tsh[c][pix] = tanhf(xb[(size_t)c * HW + pix0 + pix]);
    }
    __syncthreads();
    __hip_bfloat16* bp = basis + (size_t)pair * HW * KB;
#pragma unroll
    for (int i = 0; i < 16; ++i) {
        int idx = threadIdx.x + i * 256;
        int c = idx & 63, pix = idx >> 6;
        float tv = tsh[c][pix];
        float p0 = 1.0f;
        float p1 = tv;
        float p2 = tv * p1 - 2.25f * bw1 * p0;              // i=2: coef=9/4
        float p3 = tv * p2 - (100.0f / 3.0f) * bw2 * p1;    // i=3: coef=100/3
        float s0 = p0 / (1.0f + __expf(-p0));
        float s1 = p1 / (1.0f + __expf(-p1));
        float s2 = p2 / (1.0f + __expf(-p2));
        float s3 = p3 / (1.0f + __expf(-p3));
        __hip_bfloat16* o = bp + (size_t)(pix0 + pix) * KB + c;
        o[0]   = __float2bfloat16(s0);
        o[64]  = __float2bfloat16(s1);
        o[128] = __float2bfloat16(s2);
        o[192] = __float2bfloat16(s3);
    }
}

// ---------------------------------------------------------------------------
// 5) implicit-GEMM conv via v_wmma_f32_16x16x32_bf16 with LDS-staged B.
//    Block = (b, 16-pixel tile); 8 waves cover all 128 C_OUT rows.
//    Per expert: stage 3x18 pixel window (zero-padded halo) into LDS via
//    async global->LDS, then a branch-free 72-WMMA tap loop (B from ds reads).
// ---------------------------------------------------------------------------
__global__ void __launch_bounds__(256)
k_conv(const __hip_bfloat16* __restrict__ wpack,
       const __hip_bfloat16* __restrict__ basis,
       const int* __restrict__ tki, const float* __restrict__ tkg,
       float* __restrict__ y) {
    __shared__ __hip_bfloat16 win[3 * 18 * PIXSTR];    // 28,512 B

    int tile = blockIdx.x;                     // 16 pixels of one image row
    int b    = blockIdx.y;
    int wave = threadIdx.x >> 5;
    int lane = threadIdx.x & 31;
    int ln   = lane & 15;
    int khalf16 = (lane >= 16) ? 8 : 0;        // B-matrix K half per lane
    int yrow = tile >> 2;
    int x0   = (tile & 3) << 4;

    v8f tot = {};
    for (int kk = 0; kk < 2; ++kk) {
        int   e = tki[b * 2 + kk];
        float g = tkg[b * 2 + kk];
        const __hip_bfloat16* wp = wpack + (size_t)(e * 8 + wave) * 9 * 8 * 32 * 16;
        const __hip_bfloat16* bp = basis + (size_t)(b * 2 + kk) * HW * KB;

        __syncthreads();                       // previous readers done with win
        // ---- stage 54 pixels x 512 B into LDS, zero-padding the halo ----
        for (int u = threadIdx.x; u < 54 * 32; u += 256) {
            int pix = u >> 5, chunk = u & 31;  // chunk = 16-byte unit
            int r = pix / 18, c = pix % 18;
            int sy = yrow - 1 + r, sx = x0 - 1 + c;
            bool valid = (sy >= 0) & (sy < HH) & (sx >= 0) & (sx < WW);
            __hip_bfloat16* dst = &win[(r * 18 + c) * PIXSTR + chunk * 8];
            if (valid) {
                const __hip_bfloat16* src = bp + ((size_t)(sy * WW + sx)) * KB + chunk * 8;
#if HAS_ASYNC_LDS
                __builtin_amdgcn_global_load_async_to_lds_b128(
                    (gas_v4i*)src, (las_v4i*)dst, 0, 0);
#else
                *(float4*)dst = *(const float4*)src;
#endif
            } else {
                float4 z = {0.f, 0.f, 0.f, 0.f};
                *(float4*)dst = z;
            }
        }
#if HAS_ASYNC_LDS
#if __has_builtin(__builtin_amdgcn_s_wait_asynccnt)
        __builtin_amdgcn_s_wait_asynccnt(0);
#else
        asm volatile("s_wait_asynccnt 0" ::: "memory");
#endif
#endif
        __syncthreads();

        // ---- branch-free tap loop: 9 taps x 8 k-chunks of WMMA ----
        v8f acc = {};
        for (int tap = 0; tap < 9; ++tap) {
            int row = tap / 3, colb = tap % 3;                 // ky+1, kx+1
            const __hip_bfloat16* wtap = wp + (size_t)tap * 8 * 32 * 16;
            const __hip_bfloat16* bl = &win[(row * 18 + colb + ln) * PIXSTR + khalf16];
#pragma unroll
            for (int kc = 0; kc < 8; ++kc) {
                v16bf a = *(const v16bf*)(wtap + ((size_t)kc * 32 + lane) * 16);
                v8bf blo = *(const v8bf*)(bl + kc * 32);
                v8bf bhi = *(const v8bf*)(bl + kc * 32 + 16);
                v16bf bb = __builtin_shufflevector(blo, bhi,
                    0, 1, 2, 3, 4, 5, 6, 7, 8, 9, 10, 11, 12, 13, 14, 15);
                acc = __builtin_amdgcn_wmma_f32_16x16x32_bf16(
                        /*neg_a=*/false, a, /*neg_b=*/false, bb,
                        /*c_mod=*/(short)0, acc, /*reuse_a=*/false, /*reuse_b=*/false);
            }
        }
#pragma unroll
        for (int r = 0; r < 8; ++r) tot[r] += g * acc[r];
    }

    int mbase = wave * 16 + ((lane >= 16) ? 8 : 0);   // D-layout: VGPR r -> M=r / r+8
    float* yb = y + (size_t)b * COUT * HW + (size_t)yrow * WW;
#pragma unroll
    for (int r = 0; r < 8; ++r)
        yb[(size_t)(mbase + r) * HW + x0 + ln] = tot[r];
}

// ---------------------------------------------------------------------------
extern "C" void kernel_launch(void* const* d_in, const int* in_sizes, int n_in,
                              void* d_out, int out_size, void* d_ws, size_t ws_size,
                              hipStream_t stream) {
    const float* x    = (const float*)d_in[0];
    const float* wg   = (const float*)d_in[1];
    const float* pw   = (const float*)d_in[2];
    const float* beta = (const float*)d_in[3];
    float* y = (float*)d_out;
    char*  ws = (char*)d_ws;

    float* means = (float*)(ws + OFF_MEANS);
    float* gates = (float*)(ws + OFF_GATES);
    int*   tki   = (int*)  (ws + OFF_TKI);
    float* tkg   = (float*)(ws + OFF_TKG);
    __hip_bfloat16* wpack = (__hip_bfloat16*)(ws + OFF_WPACK);
    __hip_bfloat16* basis = (__hip_bfloat16*)(ws + OFF_BASIS);
    float* loss_out = y + (out_size - 1);

    k_mean <<<NB * CIN, 256, 0, stream>>>(x, means);
    k_gate <<<1, 64, 0, stream>>>(means, wg, gates, tki, tkg, loss_out);
    k_pack <<<72, 256, 0, stream>>>(pw, wpack);                 // 18432 lane-slots
    k_basis<<<dim3(NPAIR, HW / 64), 256, 0, stream>>>(x, beta, tki, basis);
    k_conv <<<dim3(HW / 16, NB), 256, 0, stream>>>(wpack, basis, tki, tkg, y);
}
